// MultiHeadAttn_43379169689936
// MI455X (gfx1250) — matrix operand order repaired and checked by reading, not compiled, not run on hardware
//
#include <hip/hip_runtime.h>
#include <hip/hip_bf16.h>

// ---------------------------------------------------------------------------
// MHA layer for MI455X (gfx1250): bf16 WMMA GEMMs + flash-style attention,
// TDM (tensor_load_to_lds) K-tile staging, async global->LDS copies.
// ---------------------------------------------------------------------------

typedef __attribute__((ext_vector_type(16))) __bf16 v16bf;
typedef __attribute__((ext_vector_type(8)))  __bf16 v8bf;
typedef __attribute__((ext_vector_type(4)))  __bf16 v4bf;
typedef __attribute__((ext_vector_type(8)))  float  v8f;
typedef unsigned int u32x4 __attribute__((ext_vector_type(4)));
typedef int          i32x4 __attribute__((ext_vector_type(4)));
typedef int          i32x8 __attribute__((ext_vector_type(8)));

#define B_   2
#define S_   2048
#define D_   1024
#define H_   16
#define DK_  64
#define LDA  40      // padded LDS row stride (bf16 elems) for GEMM tiles
#define KLD  72      // padded LDS row stride for TDM-staged K tiles (144 B)

#if defined(__has_builtin)
#if __has_builtin(__builtin_amdgcn_tensor_load_to_lds)
#define HAVE_TDM 1
#endif
#endif

static __device__ __forceinline__ v16bf make16(v8bf lo, v8bf hi) {
  union { v16bf v; v8bf h[2]; } u;
  u.h[0] = lo; u.h[1] = hi;
  return u.v;
}
static __device__ __forceinline__ v8f zero8() {
  v8f z = {0.f,0.f,0.f,0.f,0.f,0.f,0.f,0.f};
  return z;
}
// Low 32 bits of a generic pointer into LDS == byte offset within LDS.
static __device__ __forceinline__ unsigned lds_off(const void* p) {
  return (unsigned)(unsigned long long)p;
}

#ifdef HAVE_TDM
// TDM: DMA a 2-D bf16 tile (tile_dim0=64 contiguous elems, tile_dim1=32 rows,
// row stride 64 elems in memory) into LDS, padding each 32-DWORD row with
// 4 DWORDs so LDS rows land at a 144 B stride (bank-conflict-free b128 reads).
static __device__ __forceinline__ void tdm_load_k32x64(unsigned lds_addr,
                                                       unsigned long long gaddr) {
  u32x4 g0;
  g0[0] = 1u;                                      // count=1 (valid user D#)
  g0[1] = lds_addr;                                // lds_addr
  g0[2] = (unsigned)gaddr;                         // global_addr[31:0]
  g0[3] = (unsigned)(gaddr >> 32) | (2u << 30);    // global_addr[56:32] | type=2
  i32x8 g1;
  g1[0] = (int)((1u << 16) |                       // data_size = 2 bytes
                (1u << 20) |                       // pad_enable
                (4u << 22) |                       // pad_interval = 32 DWORDs
                (3u << 25));                       // pad_amount   = 4 DWORDs
  g1[1] = (int)(64u << 16);                        // tensor_dim0[15:0] = 64
  g1[2] = (int)(2048u << 16);                      // tensor_dim1[15:0] = 2048
  g1[3] = (int)(64u << 16);                        // tile_dim0 = 64
  g1[4] = 32;                                      // tile_dim1 = 32
  g1[5] = 64;                                      // tensor_dim0_stride = 64
  g1[6] = 0;
  g1[7] = 0;
  i32x4 z4 = {0, 0, 0, 0};
#if __clang_major__ >= 23
  i32x8 z8 = {0, 0, 0, 0, 0, 0, 0, 0};
  __builtin_amdgcn_tensor_load_to_lds(g0, g1, z4, z4, z8, 0);
#else
  __builtin_amdgcn_tensor_load_to_lds(g0, g1, z4, z4, 0);
#endif
}
#endif

// ---------------------------------------------------------------------------
// Kernel A: fused QKV projection.  C[4096,1024] = X[4096,1024] * W^T + b
// Block tile 128x128, K-step 32, 8 waves (4M x 2N), wave tile 32x64.
// Output scattered into [B,H,S,dk] bf16 for the attention kernel.
// ---------------------------------------------------------------------------
__global__ __launch_bounds__(256) void qkv_gemm(
    const float* __restrict__ Xq, const float* __restrict__ Xk, const float* __restrict__ Xv,
    const float* __restrict__ Wq, const float* __restrict__ Wk, const float* __restrict__ Wv,
    const float* __restrict__ bq, const float* __restrict__ bk, const float* __restrict__ bv,
    __bf16* __restrict__ Qo, __bf16* __restrict__ Ko, __bf16* __restrict__ Vo)
{
  const float* X; const float* W; const float* bias; __bf16* dst;
  if (blockIdx.z == 0)      { X = Xq; W = Wq; bias = bq; dst = Qo; }
  else if (blockIdx.z == 1) { X = Xk; W = Wk; bias = bk; dst = Ko; }
  else                      { X = Xv; W = Wv; bias = bv; dst = Vo; }

  __shared__ __align__(16) __bf16 As[128 * LDA];
  __shared__ __align__(16) __bf16 Bs[128 * LDA];

  const int tid   = threadIdx.x;
  const int wv    = tid >> 5;
  const int ln    = tid & 31;
  const int lh    = ln >> 4;     // lane half (0/1)
  const int lc    = ln & 15;     // lane column
  const int waveM = wv & 3;      // 4 waves along M
  const int waveN = wv >> 2;     // 2 waves along N
  const int mBase = blockIdx.y * 128;
  const int nBase = blockIdx.x * 128;

  v8f acc[2][4];
  for (int mt = 0; mt < 2; ++mt)
    for (int nt = 0; nt < 4; ++nt) acc[mt][nt] = zero8();

  for (int kb = 0; kb < D_ / 32; ++kb) {
    const int k0 = kb * 32;
    // --- stage 128x32 fp32 tiles of X and W, convert to bf16 in LDS -------
    for (int p = 0; p < 4; ++p) {
      int fid = p * 256 + tid;
      int row = fid >> 3;
      int c4  = (fid & 7) * 4;
      float4 a = *(const float4*)&X[(size_t)(mBase + row) * D_ + k0 + c4];
      float4 w = *(const float4*)&W[(size_t)(nBase + row) * D_ + k0 + c4];
      if (k0 + 32 < D_) {  // touch next K tile (global_prefetch_b8)
        __builtin_prefetch(&X[(size_t)(mBase + row) * D_ + k0 + 32 + c4], 0, 0);
      }
      v4bf ab, wb;
      ab[0] = (__bf16)a.x; ab[1] = (__bf16)a.y; ab[2] = (__bf16)a.z; ab[3] = (__bf16)a.w;
      wb[0] = (__bf16)w.x; wb[1] = (__bf16)w.y; wb[2] = (__bf16)w.z; wb[3] = (__bf16)w.w;
      *(v4bf*)&As[row * LDA + c4] = ab;
      *(v4bf*)&Bs[row * LDA + c4] = wb;
    }
    __syncthreads();

    // --- fragments + WMMA -------------------------------------------------
    v16bf af[2], bfr[4];
    for (int mt = 0; mt < 2; ++mt) {
      int r  = waveM * 32 + mt * 16 + lc;
      int c0 = lh * 8;  // A layout: halves hold K {0..7,16..23} / {8..15,24..31}
      af[mt] = make16(*(const v8bf*)&As[r * LDA + c0],
                      *(const v8bf*)&As[r * LDA + c0 + 16]);
    }
    for (int nt = 0; nt < 4; ++nt) {
      int r  = waveN * 64 + nt * 16 + lc;
      int c0 = lh * 8;  // B layout: halves hold K {0..7,16..23} / {8..15,24..31}
      bfr[nt] = make16(*(const v8bf*)&Bs[r * LDA + c0],
                       *(const v8bf*)&Bs[r * LDA + c0 + 16]);
    }
    for (int mt = 0; mt < 2; ++mt)
      for (int nt = 0; nt < 4; ++nt)
        acc[mt][nt] = __builtin_amdgcn_wmma_f32_16x16x32_bf16(
            false, af[mt], false, bfr[nt], (short)0, acc[mt][nt], false, false);
    __syncthreads();
  }

  // --- epilogue: +bias, scatter to [B,H,S,dk] bf16 -------------------------
  for (int mt = 0; mt < 2; ++mt)
    for (int nt = 0; nt < 4; ++nt)
      for (int r = 0; r < 8; ++r) {
        int grow = mBase + waveM * 32 + mt * 16 + r + lh * 8;
        int gcol = nBase + waveN * 64 + nt * 16 + lc;
        float v  = acc[mt][nt][r] + bias[gcol];
        int bb = grow >> 11;          // batch
        int ss = grow & (S_ - 1);     // seq
        int hh = gcol >> 6;           // head
        int dk = gcol & (DK_ - 1);
        dst[(((size_t)(bb * H_ + hh)) * S_ + ss) * DK_ + dk] = (__bf16)v;
      }
}

// ---------------------------------------------------------------------------
// Kernel B: flash attention.  Block = 128 q-rows of one (b,h); 8 waves x 16
// rows.  Online softmax over 32-key steps.  K chunk staged once per block by
// the Tensor Data Mover (double-buffered, pipelined with s_wait_tensorcnt);
// V staged transposed in LDS; P restaged via LDS (C-layout -> A-layout).
// ---------------------------------------------------------------------------
__global__ __launch_bounds__(256) void attn(
    const __bf16* __restrict__ Q, const __bf16* __restrict__ K,
    const __bf16* __restrict__ V, const int* __restrict__ mask,
    __bf16* __restrict__ ctx)
{
  __shared__ __align__(16) __bf16 Pb[8 * 16 * LDA]; // per-wave P tile 16x32
  __shared__ __align__(16) __bf16 Vt[DK_ * LDA];    // V^T chunk: [dk][key32]
  __shared__ __align__(16) __bf16 Kt[2][32 * KLD];  // K chunks: [key32][dk64]

  const int tid = threadIdx.x;
  const int wv  = tid >> 5;
  const int ln  = tid & 31;
  const int lh  = ln >> 4;
  const int lc  = ln & 15;
  const int b   = blockIdx.z;
  const int h   = blockIdx.y;
  const int qBase = blockIdx.x * 128;

  const __bf16* Qh = Q + (size_t)(b * H_ + h) * S_ * DK_;
  const __bf16* Kh = K + (size_t)(b * H_ + h) * S_ * DK_;
  const __bf16* Vh = V + (size_t)(b * H_ + h) * S_ * DK_;
  const int*    mp = mask + (size_t)b * S_ * S_;

  // Q A-fragments (16 rows x dk64 -> two 16x32 frags), loaded once.
  v16bf aQ[2];
  {
    int qrow = qBase + wv * 16 + lc;
    for (int kk = 0; kk < 2; ++kk) {
      int c0 = kk * 32 + lh * 8;
      aQ[kk] = make16(*(const v8bf*)&Qh[(size_t)qrow * DK_ + c0],
                      *(const v8bf*)&Qh[(size_t)qrow * DK_ + c0 + 16]);
    }
  }

  float mrow[8], lrow[8];
  v8f acc[4];
  for (int r = 0; r < 8; ++r) { mrow[r] = -1e30f; lrow[r] = 0.f; }
  for (int n = 0; n < 4; ++n) acc[n] = zero8();

#ifdef HAVE_TDM
  if (wv == 0) {  // prologue: DMA first K chunk into buffer 0
    tdm_load_k32x64(lds_off(&Kt[0][0]), (unsigned long long)(size_t)Kh);
  }
#endif

  for (int kb = 0; kb < S_ / 32; ++kb) {
    // --- cooperative: stage V chunk transposed [dk][key] ------------------
    {
      int key = tid >> 3;
      int c8  = (tid & 7) * 8;
      v8bf vv = *(const v8bf*)&Vh[(size_t)(kb * 32 + key) * DK_ + c8];
      for (int e = 0; e < 8; ++e) Vt[(c8 + e) * LDA + key] = vv[e];
    }
#ifdef HAVE_TDM
    const __bf16* Kc = &Kt[kb & 1][0];
    if (wv == 0) {
      if (kb + 1 < S_ / 32) {  // issue next tile, then wait for current only
        tdm_load_k32x64(lds_off(&Kt[(kb + 1) & 1][0]),
                        (unsigned long long)(size_t)(Kh + (size_t)(kb + 1) * 32 * DK_));
        __builtin_amdgcn_s_wait_tensorcnt(1);
      } else {
        __builtin_amdgcn_s_wait_tensorcnt(0);
      }
    }
#else
    const __bf16* Kc = &Kt[0][0];
    {
      int row = tid >> 3;
      int c8  = (tid & 7) * 8;
      *(v8bf*)&Kt[0][row * KLD + c8] =
          *(const v8bf*)&Kh[(size_t)(kb * 32 + row) * DK_ + c8];
    }
#endif
    __syncthreads();

    // --- scores: two 16x16 tiles over 32 keys (K frags from LDS) ----------
    float xs[2][8];
    for (int nt = 0; nt < 2; ++nt) {
      v8f sc = zero8();
      int keyl = nt * 16 + lc;          // local key row in chunk
      for (int kk = 0; kk < 2; ++kk) {
        int c0 = kk * 32 + lh * 8;
        v16bf bK = make16(*(const v8bf*)&Kc[keyl * KLD + c0],
                          *(const v8bf*)&Kc[keyl * KLD + c0 + 16]);
        sc = __builtin_amdgcn_wmma_f32_16x16x32_bf16(
            false, aQ[kk], false, bK, (short)0, sc, false, false);
      }
      int keyc = kb * 32 + keyl;
      for (int r = 0; r < 8; ++r) {
        int qr  = qBase + wv * 16 + r + lh * 8;
        int msk = mp[(size_t)qr * S_ + keyc];
        xs[nt][r] = msk ? sc[r] * 0.125f : -1e10f;   // 1/sqrt(64), mask->NEG
      }
    }

    // --- online softmax (row stats uniform across each 16-lane half) ------
    for (int r = 0; r < 8; ++r) {
      float rm = fmaxf(xs[0][r], xs[1][r]);
      for (int d = 1; d < 16; d <<= 1) rm = fmaxf(rm, __shfl_xor(rm, d, 32));
      float mn = fmaxf(mrow[r], rm);
      float al = __expf(mrow[r] - mn);
      float p0 = __expf(xs[0][r] - mn);
      float p1 = __expf(xs[1][r] - mn);
      float rs = p0 + p1;
      for (int d = 1; d < 16; d <<= 1) rs += __shfl_xor(rs, d, 32);
      lrow[r] = lrow[r] * al + rs;
      mrow[r] = mn;
      xs[0][r] = p0; xs[1][r] = p1;
      for (int n = 0; n < 4; ++n) acc[n][r] *= al;
    }

    // --- P: C-layout -> LDS -> A-layout -----------------------------------
    __bf16* pb = &Pb[wv * 16 * LDA];
    for (int nt = 0; nt < 2; ++nt)
      for (int r = 0; r < 8; ++r)
        pb[(r + lh * 8) * LDA + nt * 16 + lc] = (__bf16)xs[nt][r];
    __syncthreads();

    // --- PV: ctx += P(16x32) * V(32x64) -----------------------------------
    {
      int c0 = lh * 8;
      v16bf aP = make16(*(const v8bf*)&pb[lc * LDA + c0],
                        *(const v8bf*)&pb[lc * LDA + c0 + 16]);
      for (int n = 0; n < 4; ++n) {
        int dc = n * 16 + lc;
        int ko = lh * 8;
        v16bf bV = make16(*(const v8bf*)&Vt[dc * LDA + ko],
                          *(const v8bf*)&Vt[dc * LDA + ko + 16]);
        acc[n] = __builtin_amdgcn_wmma_f32_16x16x32_bf16(
            false, aP, false, bV, (short)0, acc[n], false, false);
      }
    }
    __syncthreads();
  }

  // --- normalize and store ctx as [B,S,D] bf16 -----------------------------
  for (int n = 0; n < 4; ++n)
    for (int r = 0; r < 8; ++r) {
      int qr   = qBase + wv * 16 + r + lh * 8;
      int dcol = h * DK_ + n * 16 + lc;
      ctx[((size_t)(b * S_ + qr)) * D_ + dcol] = (__bf16)(acc[n][r] / lrow[r]);
    }
}

// ---------------------------------------------------------------------------
// Kernel C1: output projection + bias + residual -> fp32 pre-LN activations.
// The bf16 A-tile needs no conversion, so it is copied global->LDS with the
// async-DMA path (global_load_async_to_lds_b128, ASYNCcnt-tracked).
// ---------------------------------------------------------------------------
__global__ __launch_bounds__(256) void oproj(
    const __bf16* __restrict__ Xc, const float* __restrict__ Wo,
    const float* __restrict__ bo, const float* __restrict__ resid,
    float* __restrict__ xout)
{
  __shared__ __align__(16) __bf16 As[128 * LDA];
  __shared__ __align__(16) __bf16 Bs[128 * LDA];

  const int tid   = threadIdx.x;
  const int wv    = tid >> 5;
  const int ln    = tid & 31;
  const int lh    = ln >> 4;
  const int lc    = ln & 15;
  const int waveM = wv & 3;
  const int waveN = wv >> 2;
  const int mBase = blockIdx.y * 128;
  const int nBase = blockIdx.x * 128;

  v8f acc[2][4];
  for (int mt = 0; mt < 2; ++mt)
    for (int nt = 0; nt < 4; ++nt) acc[mt][nt] = zero8();

  for (int kb = 0; kb < D_ / 32; ++kb) {
    const int k0 = kb * 32;
    // A tile already bf16: async DMA straight into LDS, no VGPR round-trip.
    for (int p = 0; p < 2; ++p) {
      int fid = p * 256 + tid;
      int row = fid >> 2;
      int c8  = (fid & 3) * 8;
      unsigned           la = lds_off(&As[row * LDA + c8]);
      unsigned long long ga =
          (unsigned long long)(size_t)&Xc[(size_t)(mBase + row) * D_ + k0 + c8];
      asm volatile("global_load_async_to_lds_b128 %0, %1, off"
                   :: "v"(la), "v"(ga) : "memory");
    }
    // W tile fp32 -> bf16 through VGPRs.
    for (int p = 0; p < 4; ++p) {
      int fid = p * 256 + tid;
      int row = fid >> 3;
      int c4  = (fid & 7) * 4;
      float4 w = *(const float4*)&Wo[(size_t)(nBase + row) * D_ + k0 + c4];
      v4bf wb;
      wb[0] = (__bf16)w.x; wb[1] = (__bf16)w.y; wb[2] = (__bf16)w.z; wb[3] = (__bf16)w.w;
      *(v4bf*)&Bs[row * LDA + c4] = wb;
    }
    asm volatile("s_wait_asynccnt 0x0" ::: "memory");
    __syncthreads();

    v16bf af[2], bfr[4];
    for (int mt = 0; mt < 2; ++mt) {
      int r  = waveM * 32 + mt * 16 + lc;
      int c0 = lh * 8;
      af[mt] = make16(*(const v8bf*)&As[r * LDA + c0],
                      *(const v8bf*)&As[r * LDA + c0 + 16]);
    }
    for (int nt = 0; nt < 4; ++nt) {
      int r  = waveN * 64 + nt * 16 + lc;
      int c0 = lh * 8;
      bfr[nt] = make16(*(const v8bf*)&Bs[r * LDA + c0],
                       *(const v8bf*)&Bs[r * LDA + c0 + 16]);
    }
    for (int mt = 0; mt < 2; ++mt)
      for (int nt = 0; nt < 4; ++nt)
        acc[mt][nt] = __builtin_amdgcn_wmma_f32_16x16x32_bf16(
            false, af[mt], false, bfr[nt], (short)0, acc[mt][nt], false, false);
    __syncthreads();
  }

  for (int mt = 0; mt < 2; ++mt)
    for (int nt = 0; nt < 4; ++nt)
      for (int r = 0; r < 8; ++r) {
        int grow = mBase + waveM * 32 + mt * 16 + r + lh * 8;
        int gcol = nBase + waveN * 64 + nt * 16 + lc;
        size_t idx = (size_t)grow * D_ + gcol;
        xout[idx] = acc[mt][nt][r] + bo[gcol] + resid[idx];
      }
}

// ---------------------------------------------------------------------------
// Kernel C2: LayerNorm over D=1024, one block per row.
// ---------------------------------------------------------------------------
__global__ __launch_bounds__(256) void lnorm(
    const float* __restrict__ x, const float* __restrict__ gamma,
    const float* __restrict__ beta, float* __restrict__ out)
{
  __shared__ float red[16];
  const int row = blockIdx.x;
  const int tid = threadIdx.x;
  const float* xr = x + (size_t)row * D_;

  float4 v = *(const float4*)&xr[tid * 4];
  float s = v.x + v.y + v.z + v.w;
  float q = v.x * v.x + v.y * v.y + v.z * v.z + v.w * v.w;
  for (int d = 1; d < 32; d <<= 1) {
    s += __shfl_xor(s, d, 32);
    q += __shfl_xor(q, d, 32);
  }
  int wv = tid >> 5, lane = tid & 31;
  if (lane == 0) { red[wv] = s; red[8 + wv] = q; }
  __syncthreads();
  float st = 0.f, qt = 0.f;
  for (int i = 0; i < 8; ++i) { st += red[i]; qt += red[8 + i]; }
  float mean = st * (1.f / (float)D_);
  float var  = qt * (1.f / (float)D_) - mean * mean;
  float rstd = rsqrtf(var + 1e-6f);

  float4 g  = *(const float4*)&gamma[tid * 4];
  float4 be = *(const float4*)&beta[tid * 4];
  float4 o;
  o.x = (v.x - mean) * rstd * g.x + be.x;
  o.y = (v.y - mean) * rstd * g.y + be.y;
  o.z = (v.z - mean) * rstd * g.z + be.z;
  o.w = (v.w - mean) * rstd * g.w + be.w;
  *(float4*)&out[(size_t)row * D_ + tid * 4] = o;
}

// ---------------------------------------------------------------------------
extern "C" void kernel_launch(void* const* d_in, const int* in_sizes, int n_in,
                              void* d_out, int out_size, void* d_ws, size_t ws_size,
                              hipStream_t stream) {
  const float* query  = (const float*)d_in[0];
  const float* key    = (const float*)d_in[1];
  const float* values = (const float*)d_in[2];
  const int*   mask   = (const int*)d_in[3];
  const float* Wq = (const float*)d_in[4];  const float* bq = (const float*)d_in[5];
  const float* Wk = (const float*)d_in[6];  const float* bk = (const float*)d_in[7];
  const float* Wv = (const float*)d_in[8];  const float* bv = (const float*)d_in[9];
  const float* Wo = (const float*)d_in[10]; const float* bo = (const float*)d_in[11];
  const float* gamma = (const float*)d_in[12];
  const float* beta  = (const float*)d_in[13];

  char* ws = (char*)d_ws;
  const size_t MB = 1024 * 1024;
  __bf16* Qb = (__bf16*)(ws + 0 * MB);   // [B,H,S,dk] bf16, 8 MB
  __bf16* Kb = (__bf16*)(ws + 8 * MB);   // 8 MB
  __bf16* Vb = (__bf16*)(ws + 16 * MB);  // 8 MB
  __bf16* Cb = (__bf16*)(ws + 24 * MB);  // ctx [B,S,D] bf16, 8 MB
  float*  Xb = (float*)(ws + 32 * MB);   // pre-LN fp32, 16 MB

  qkv_gemm<<<dim3(D_ / 128, (B_ * S_) / 128, 3), 256, 0, stream>>>(
      query, key, values, Wq, Wk, Wv, bq, bk, bv, Qb, Kb, Vb);
  attn<<<dim3(S_ / 128, H_, B_), 256, 0, stream>>>(Qb, Kb, Vb, mask, Cb);
  oproj<<<dim3(D_ / 128, (B_ * S_) / 128), 256, 0, stream>>>(Cb, Wo, bo, query, Xb);
  lnorm<<<B_ * S_, 256, 0, stream>>>(Xb, gamma, beta, (float*)d_out);
}
